// ResLSTM_36799279792503
// MI455X (gfx1250) — matrix operand order repaired and checked by reading, not compiled, not run on hardware
//
#include <hip/hip_runtime.h>
#include <math.h>

typedef __bf16 bf16_t;
typedef __attribute__((ext_vector_type(16))) __bf16 v16bf;
typedef __attribute__((ext_vector_type(8)))  __bf16 v8bf;
typedef __attribute__((ext_vector_type(8)))  float  v8f;
typedef __attribute__((ext_vector_type(4)))  unsigned int v4u;
typedef __attribute__((ext_vector_type(4)))  int v4i;
typedef __attribute__((ext_vector_type(8)))  int v8i;

union U16bf { v16bf v; v8bf h[2]; };

#define L_NUM 2
#define H_DIM 512
#define T_SEQ 1024
#define B_SZ  64
#define G3H   1536
#define M_TOT (T_SEQ * B_SZ)
#define REC_BLOCKS 28
#define REC_THREADS 256
#define SMEM_BYTES 65536   // 64 x 512 bf16 A-matrix staged in LDS

#if __has_builtin(__builtin_amdgcn_tensor_load_to_lds)
#define USE_TDM 1
#else
#define USE_TDM 0
#endif

// ---------------- workspace layout (bytes) ----------------
static constexpr size_t SZ_WG   = (size_t)L_NUM * G3H * H_DIM;    // Wii/Wih/Wic elem count
static constexpr size_t SZ_WHH  = (size_t)L_NUM * H_DIM * H_DIM;
static constexpr size_t OFF_WII   = 0;
static constexpr size_t OFF_WIH   = OFF_WII + SZ_WG * 2;
static constexpr size_t OFF_WIC   = OFF_WIH + SZ_WG * 2;
static constexpr size_t OFF_WHH   = OFF_WIC + SZ_WG * 2;
static constexpr size_t OFF_BIASG = OFF_WHH + SZ_WHH * 2;                 // f32 [L][3H]
static constexpr size_t OFF_CF    = OFF_BIASG + (size_t)L_NUM * G3H * 4;  // f32 cell state
static constexpr size_t OFF_HB    = OFF_CF + (size_t)B_SZ * H_DIM * 4;    // bf16 h
static constexpr size_t OFF_CB    = OFF_HB + (size_t)B_SZ * H_DIM * 2;    // bf16 c
static constexpr size_t OFF_BAR   = OFF_CB + (size_t)B_SZ * H_DIM * 2;    // barrier cnt/gen
static constexpr size_t OFF_GH    = OFF_BAR + 256;                        // f32 [64][3H]
static constexpr size_t OFF_GC    = OFF_GH + (size_t)B_SZ * G3H * 4;      // f32 [64][3H]
static constexpr size_t OFF_PRE   = OFF_GC + (size_t)B_SZ * G3H * 4;      // f32 [64][H]
static constexpr size_t OFF_XIN   = OFF_PRE + (size_t)B_SZ * H_DIM * 4;   // bf16 [T*B][H]
static constexpr size_t OFF_XPROJ = OFF_XIN + (size_t)M_TOT * H_DIM * 2;  // bf16 [T*B][3H]
static constexpr long   STATE_WORDS = (long)((OFF_GH - OFF_CF) / 4);

// ---------------- device helpers ----------------
static __device__ __forceinline__ v8f wmma_bf16(v16bf a, v16bf b, v8f c) {
  return __builtin_amdgcn_wmma_f32_16x16x32_bf16(false, a, false, b, (short)0, c,
                                                 false, false);
}

static __device__ __forceinline__ float sigmoidf_(float x) {
  return 1.0f / (1.0f + __expf(-x));
}

// A: 64 x H_DIM bf16 row-major tile base (global or LDS); Brow: one row
// (length H_DIM) of row-major weight W[N][K] selected by this lane's column.
// acc[mt] += A[mt*16 .. +16, :] @ W_row^T (16x16 f32 accum per M-tile).
// B fragment is software-pipelined one k-tile ahead so WMMA does not have to
// drain all outstanding loads (avoids s_wait_loadcnt 0 serialization).
static __device__ __forceinline__ void gemm64_tile(const bf16_t* __restrict__ A,
                                                   const bf16_t* __restrict__ Brow,
                                                   int nl, int hl, v8f acc[4]) {
  U16bf bcur;
  bcur.h[0] = *(const v8bf*)(Brow + hl * 8);
  bcur.h[1] = *(const v8bf*)(Brow + 16 + hl * 8);
  U16bf bnxt = bcur;
  for (int kt = 0; kt < 16; ++kt) {       // K = 512 = 16 * 32
    const int k0 = kt * 32;
    if (kt < 15) {
      bnxt.h[0] = *(const v8bf*)(Brow + k0 + 32 + hl * 8);
      bnxt.h[1] = *(const v8bf*)(Brow + k0 + 32 + 16 + hl * 8);
    }
#pragma unroll
    for (int mt = 0; mt < 4; ++mt) {
      const bf16_t* Ar = A + (size_t)(mt * 16 + nl) * H_DIM;  // row = lane&15
      U16bf af;
      // A fragment: elems 0-7 -> K = k0 + hl*8 + e ; elems 8-15 -> +16
      af.h[0] = *(const v8bf*)(Ar + k0 + hl * 8);
      af.h[1] = *(const v8bf*)(Ar + k0 + 16 + hl * 8);
      acc[mt] = wmma_bf16(af.v, bcur.v, acc[mt]);
    }
    bcur = bnxt;
  }
}

// TDM: DMA 64KB (64 x 512 bf16) from global into LDS. 1-D descriptor:
// data_size=8B (code 3), tensor_dim0 = tile_dim0 = 8192 elems, dim1 = 1.
// Group0: count=1 | lds_addr | global_addr[56:0] | type=2.  (ISA 8.3/8.4)
static __device__ __forceinline__ void tdm_stage_64k(const void* gsrc,
                                                     unsigned lds_off) {
#if USE_TDM
  unsigned long long ga = (unsigned long long)gsrc;
  v4u g0 = { 1u,                                   // count = 1 (user descriptor)
             lds_off,                              // LDS byte address
             (unsigned)ga,                         // global_addr[31:0]
             ((unsigned)(ga >> 32) & 0x01FFFFFFu) | 0x80000000u };  // type=2
  v8i g1 = { 0x00030000,        // workgroup_mask=0, data_size=3 (8 bytes)
             0x20000000,        // tensor_dim0[15:0]=8192 in bits[31:16]
             0x00010000,        // tensor_dim0[31:16]=0, tensor_dim1=1
             0x20000000,        // tile_dim0 = 8192 in bits[31:16]
             0x00000001,        // tile_dim1 = 1, tile_dim2 = 0
             0x00002000,        // tensor_dim0_stride = 8192
             0, 0 };
  v4i gz = { 0, 0, 0, 0 };
#if defined(__clang_major__) && (__clang_major__ >= 23)
  v8i gz8 = { 0, 0, 0, 0, 0, 0, 0, 0 };
  __builtin_amdgcn_tensor_load_to_lds(g0, g1, gz, gz, gz8, 0);
#else
  __builtin_amdgcn_tensor_load_to_lds(g0, g1, gz, gz, 0);
#endif
#endif
}

// Monotonic device-wide sense barrier (no counter reset race).
static __device__ __forceinline__ void gbar(unsigned* cnt, unsigned* gen,
                                            unsigned& epoch) {
  __threadfence();
  __syncthreads();
  if (threadIdx.x == 0) {
    ++epoch;
    unsigned arrive =
        __hip_atomic_fetch_add(cnt, 1u, __ATOMIC_ACQ_REL, __HIP_MEMORY_SCOPE_AGENT);
    if (arrive == epoch * (unsigned)REC_BLOCKS - 1u) {
      __hip_atomic_store(gen, epoch, __ATOMIC_RELEASE, __HIP_MEMORY_SCOPE_AGENT);
    } else {
      while (__hip_atomic_load(gen, __ATOMIC_ACQUIRE, __HIP_MEMORY_SCOPE_AGENT) <
             epoch) {
        __builtin_amdgcn_s_sleep(2);
      }
    }
  }
  __syncthreads();
}

// ---------------- utility kernels ----------------
__global__ void k_cvt(bf16_t* dst, const float* src, long n) {
  long i = (long)blockIdx.x * blockDim.x + threadIdx.x;
  if (i < n) dst[i] = (bf16_t)src[i];
}

__global__ void k_biassum(float* dst, const float* a, const float* b,
                          const float* c, int n) {
  int i = blockIdx.x * blockDim.x + threadIdx.x;
  if (i < n) dst[i] = a[i] + b[i] + c[i];
}

__global__ void k_zero(unsigned* p, long n) {
  long i = (long)blockIdx.x * blockDim.x + threadIdx.x;
  if (i < n) p[i] = 0u;
}

// ---------------- parallel input projection ----------------
// xproj[m, n] = sum_k xin[m,k] * Wii[n,k] + (bii+bic+bih)[n], stored bf16.
// M = 65536, N = 1536, K = 512. One wave per 64x16 output strip; latency
// hidden by 98304 independent wave-tasks.
__global__ __launch_bounds__(256) void k_xproj(const bf16_t* __restrict__ xin,
                                               const bf16_t* __restrict__ wii_l,
                                               const float* __restrict__ biasg_l,
                                               bf16_t* __restrict__ xproj) {
  const int lane = threadIdx.x & 31;
  const int wave = threadIdx.x >> 5;
  const int task = blockIdx.x * 8 + wave;   // 0 .. 98303
  const int mstrip = task / 96;             // 0 .. 1023 (64 rows each)
  const int nt = task - mstrip * 96;        // 0 .. 95
  const int nl = lane & 15, hl = lane >> 4;
  const int n = nt * 16 + nl;

  const float bias = biasg_l[n];
  v8f acc[4];
#pragma unroll
  for (int mt = 0; mt < 4; ++mt)
#pragma unroll
    for (int v = 0; v < 8; ++v) acc[mt][v] = bias;

  gemm64_tile(xin + (size_t)mstrip * 64 * H_DIM, wii_l + (size_t)n * H_DIM, nl, hl,
              acc);

#pragma unroll
  for (int mt = 0; mt < 4; ++mt)
#pragma unroll
    for (int v = 0; v < 8; ++v) {
      const size_t m = (size_t)mstrip * 64 + mt * 16 + hl * 8 + v;
      xproj[m * G3H + n] = (bf16_t)acc[mt][v];
    }
}

// ---------------- persistent recurrent scan ----------------
// 28 blocks x 8 waves = 224 waves; role-homogeneous blocks:
//   blocks  0-11 (waves   0- 95): Gh  = xproj[t] + h @ Wih^T
//   blocks 12-23 (waves  96-191): Gc  = c @ Wic^T
//   blocks 24-27 (waves 192-223): Pre = h @ Whh^T
// Per step each block TDM-stages its 64KB A-matrix (h or c, bf16) into LDS,
// all 8 waves consume it via ds loads; device barrier; elementwise update.
__global__ __launch_bounds__(256) void k_recur(
    const bf16_t* __restrict__ wih_l, const bf16_t* __restrict__ wic_l,
    const bf16_t* __restrict__ whh_l, const bf16_t* __restrict__ xproj,
    const float* __restrict__ bhh_l, const float* __restrict__ xres,
    float* __restrict__ outs, float* __restrict__ hn_l, float* __restrict__ cn_l,
    float* __restrict__ cF, bf16_t* __restrict__ hB, bf16_t* __restrict__ cB,
    float* __restrict__ Gh, float* __restrict__ Gc, float* __restrict__ Pre,
    bf16_t* xin_next, unsigned* barCnt, unsigned* barGen) {
  extern __shared__ char smem[];
  bf16_t* sA = (bf16_t*)smem;
  const unsigned lds_off = (unsigned)(unsigned long long)smem;  // LDS byte addr

  const int lane = threadIdx.x & 31;
  const int wave = threadIdx.x >> 5;
  const int gw = blockIdx.x * 8 + wave;     // 0 .. 223
  const int nl = lane & 15, hl = lane >> 4;
  const int tid = blockIdx.x * REC_THREADS + threadIdx.x;  // 0 .. 7167
  const bool isCblk = (blockIdx.x >= 12 && blockIdx.x < 24);
  const bf16_t* Asrc = isCblk ? cB : hB;    // which state matrix this block uses
  unsigned epoch = 0;

  for (int t = 0; t < T_SEQ; ++t) {
    // ---- stage A-matrix (64 x 512 bf16) into LDS via Tensor Data Mover ----
#if USE_TDM
    if (threadIdx.x < 32) {                 // wave 0 issues the DMA
      tdm_stage_64k(Asrc, lds_off);
#if __has_builtin(__builtin_amdgcn_s_wait_tensorcnt)
      __builtin_amdgcn_s_wait_tensorcnt(0);
#endif
    }
#else
    {
      const uint4* gs = (const uint4*)Asrc;
      uint4* ds = (uint4*)smem;
      for (int i = threadIdx.x; i < SMEM_BYTES / 16; i += REC_THREADS)
        ds[i] = gs[i];
    }
#endif
    __syncthreads();

    const bf16_t* xp_t = xproj + (size_t)t * B_SZ * G3H;

    // ---- phase 1: recurrent GEMMs (wave-uniform roles, EXEC all ones) ----
    if (gw < 96) {                                   // gates from h (+ xproj init)
      const int n = gw * 16 + nl;
      v8f acc[4];
#pragma unroll
      for (int mt = 0; mt < 4; ++mt)
#pragma unroll
        for (int v = 0; v < 8; ++v)
          acc[mt][v] = (float)xp_t[(size_t)(mt * 16 + hl * 8 + v) * G3H + n];
      gemm64_tile(sA, wih_l + (size_t)n * H_DIM, nl, hl, acc);
#pragma unroll
      for (int mt = 0; mt < 4; ++mt)
#pragma unroll
        for (int v = 0; v < 8; ++v)
          Gh[(size_t)(mt * 16 + hl * 8 + v) * G3H + n] = acc[mt][v];
    } else if (gw < 192) {                           // gates from c
      const int n = (gw - 96) * 16 + nl;
      v8f acc[4] = {};
      gemm64_tile(sA, wic_l + (size_t)n * H_DIM, nl, hl, acc);
#pragma unroll
      for (int mt = 0; mt < 4; ++mt)
#pragma unroll
        for (int v = 0; v < 8; ++v)
          Gc[(size_t)(mt * 16 + hl * 8 + v) * G3H + n] = acc[mt][v];
    } else {                                         // cellgate pre-activation
      const int n = (gw - 192) * 16 + nl;
      v8f acc[4] = {};
      gemm64_tile(sA, whh_l + (size_t)n * H_DIM, nl, hl, acc);
#pragma unroll
      for (int mt = 0; mt < 4; ++mt)
#pragma unroll
        for (int v = 0; v < 8; ++v)
          Pre[(size_t)(mt * 16 + hl * 8 + v) * H_DIM + n] = acc[mt][v];
    }

    gbar(barCnt, barGen, epoch);

    // ---- phase 2: elementwise LSTM cell update ----
    for (int idx = tid; idx < B_SZ * H_DIM; idx += REC_BLOCKS * REC_THREADS) {
      const int b = idx >> 9;
      const int n = idx & (H_DIM - 1);
      const size_t gb = (size_t)b * G3H + n;
      const float gi = Gh[gb] + Gc[gb];
      const float gf = Gh[gb + H_DIM] + Gc[gb + H_DIM];
      const float go = Gh[gb + 2 * H_DIM] + Gc[gb + 2 * H_DIM];
      const float cg = tanhf(Pre[idx] + bhh_l[n]);
      const float cold = cF[idx];
      const float cy = sigmoidf_(gf) * cold + sigmoidf_(gi) * cg;
      const float xt = xres[(size_t)(t * B_SZ + b) * H_DIM + n];
      const float hy = sigmoidf_(go) * (tanhf(cy) + xt);  // residual
      cF[idx] = cy;
      cB[idx] = (bf16_t)cy;
      hB[idx] = (bf16_t)hy;
      outs[(size_t)(t * B_SZ + b) * H_DIM + n] = hy;
      if (xin_next) xin_next[(size_t)(t * B_SZ + b) * H_DIM + n] = (bf16_t)hy;
      if (t == T_SEQ - 1) {
        hn_l[idx] = hy;
        cn_l[idx] = cy;
      }
    }

    gbar(barCnt, barGen, epoch);
  }
}

// ---------------- host launcher ----------------
static inline unsigned cdiv(long a, long b) { return (unsigned)((a + b - 1) / b); }

extern "C" void kernel_launch(void* const* d_in, const int* in_sizes, int n_in,
                              void* d_out, int out_size, void* d_ws, size_t ws_size,
                              hipStream_t stream) {
  const float* x   = (const float*)d_in[0];
  const float* Wii = (const float*)d_in[1];
  const float* Wic = (const float*)d_in[2];
  const float* Wih = (const float*)d_in[3];
  const float* bii = (const float*)d_in[4];
  const float* bic = (const float*)d_in[5];
  const float* bih = (const float*)d_in[6];
  const float* Whh = (const float*)d_in[7];
  const float* bhh = (const float*)d_in[8];

  char* ws = (char*)d_ws;
  bf16_t* wiiB   = (bf16_t*)(ws + OFF_WII);
  bf16_t* wihB   = (bf16_t*)(ws + OFF_WIH);
  bf16_t* wicB   = (bf16_t*)(ws + OFF_WIC);
  bf16_t* whhB   = (bf16_t*)(ws + OFF_WHH);
  float*  biasg  = (float*)(ws + OFF_BIASG);
  bf16_t* xinB   = (bf16_t*)(ws + OFF_XIN);
  bf16_t* xprojB = (bf16_t*)(ws + OFF_XPROJ);

  float* outs = (float*)d_out;
  float* hn = outs + (size_t)T_SEQ * B_SZ * H_DIM;
  float* cn = hn + (size_t)L_NUM * B_SZ * H_DIM;

  const long nWg = (long)SZ_WG;
  k_cvt<<<cdiv(nWg, 256), 256, 0, stream>>>(wiiB, Wii, nWg);
  k_cvt<<<cdiv(nWg, 256), 256, 0, stream>>>(wihB, Wih, nWg);
  k_cvt<<<cdiv(nWg, 256), 256, 0, stream>>>(wicB, Wic, nWg);
  const long nWh = (long)SZ_WHH;
  k_cvt<<<cdiv(nWh, 256), 256, 0, stream>>>(whhB, Whh, nWh);
  k_biassum<<<cdiv(L_NUM * G3H, 256), 256, 0, stream>>>(biasg, bii, bic, bih,
                                                        L_NUM * G3H);
  const long nX = (long)M_TOT * H_DIM;
  k_cvt<<<cdiv(nX, 256), 256, 0, stream>>>(xinB, x, nX);

  for (int l = 0; l < L_NUM; ++l) {
    // parallel x-projection GEMM (65536 x 1536 x 512, bf16 WMMA)
    k_xproj<<<(M_TOT / 64) * (G3H / 16) / 8, 256, 0, stream>>>(
        xinB, wiiB + (size_t)l * G3H * H_DIM, biasg + (size_t)l * G3H, xprojB);
    // zero h/c state + barrier counters
    k_zero<<<cdiv(STATE_WORDS, 256), 256, 0, stream>>>((unsigned*)(ws + OFF_CF),
                                                       STATE_WORDS);
    // persistent recurrent scan (64KB dynamic LDS for the staged A-matrix)
    k_recur<<<REC_BLOCKS, REC_THREADS, SMEM_BYTES, stream>>>(
        wihB + (size_t)l * G3H * H_DIM, wicB + (size_t)l * G3H * H_DIM,
        whhB + (size_t)l * H_DIM * H_DIM, xprojB, bhh + (size_t)l * H_DIM,
        (l == 0) ? x : outs, outs, hn + (size_t)l * B_SZ * H_DIM,
        cn + (size_t)l * B_SZ * H_DIM, (float*)(ws + OFF_CF),
        (bf16_t*)(ws + OFF_HB), (bf16_t*)(ws + OFF_CB), (float*)(ws + OFF_GH),
        (float*)(ws + OFF_GC), (float*)(ws + OFF_PRE),
        (l == 0) ? xinB : (bf16_t*)nullptr, (unsigned*)(ws + OFF_BAR),
        (unsigned*)(ws + OFF_BAR) + 1);
  }
  (void)in_sizes; (void)n_in; (void)out_size; (void)ws_size;
}
